// S5Dual_89653147337059
// MI455X (gfx1250) — matrix operand order repaired and checked by reading, not compiled, not run on hardware
//
#include <hip/hip_runtime.h>

typedef _Float16 v16h __attribute__((ext_vector_type(16)));
typedef _Float16 v8h  __attribute__((ext_vector_type(8)));
typedef _Float16 h2   __attribute__((ext_vector_type(2)));
typedef float    v8f  __attribute__((ext_vector_type(8)));
typedef float    v4f  __attribute__((ext_vector_type(4)));

#define BATCH 4
#define SEQL  8192
#define HDIM  512
#define PDIM  256
#define BL    (BATCH * SEQL)   /* 32768 rows */
#define K2P   (2 * PDIM)       /* 512: interleaved re/im (k = 2p+ri) */
#define ROWP  (HDIM + 8)       /* LDS row pitch in halfs: +16B skew -> conflict-free b128 */
#define CHUNK 512
#define NCHUNK (SEQL / CHUNK)  /* 16 */
#define PFD   24               /* scan prefetch distance (rows) */

/* ---------------- workspace layout (bytes) ---------------- */
#define OFF_X16   ((size_t)0)                         /* BL*HDIM f16      = 33,554,432 */
#define OFF_BU    ((size_t)33554432)                  /* 2*BL*K2P f16     = 67,108,864 */
#define OFF_BB    ((size_t)100663296)                 /* 2*K2P*HDIM f16   =  1,048,576 */
#define OFF_CT    ((size_t)101711872)                 /* 2*HDIM*K2P f16   =  1,048,576 */
#define OFF_ABAR  ((size_t)102760448)                 /* 2*2*PDIM f32     =      4,096 */
#define OFF_W     ((size_t)102764544)                 /* 2*2*PDIM f32     =      4,096 */
#define OFF_DSUM  ((size_t)102768640)                 /* HDIM f32         =      2,048 */
#define OFF_SCAN  ((size_t)102770688)                 /* 4 * 32768 f32    =    524,288 */

/* ============ parameter prep: a_bar and w = (a_bar-1)/lambda ============ */
__global__ void s5_params(const float* __restrict__ lre, const float* __restrict__ lim,
                          const float* __restrict__ ldt,
                          float* __restrict__ abar, float* __restrict__ wbuf) {
  int t = blockIdx.x * blockDim.x + threadIdx.x;          /* 0..511 */
  if (t >= 2 * PDIM) return;
  int dir = t / PDIM, p = t % PDIM;
  float lr = lre[t], li = lim[t];
  float dt = __expf(ldt[t]);
  float mag = __expf(lr * dt);
  float ar = mag * __cosf(li * dt);
  float ai = mag * __sinf(li * dt);
  float den = lr * lr + li * li;
  float wr = ((ar - 1.0f) * lr + ai * li) / den;
  float wi = (ai * lr - (ar - 1.0f) * li) / den;
  abar[(dir * 2 + 0) * PDIM + p] = ar;
  abar[(dir * 2 + 1) * PDIM + p] = ai;
  wbuf[(dir * 2 + 0) * PDIM + p] = wr;
  wbuf[(dir * 2 + 1) * PDIM + p] = wi;
}

/* ============ b_bar -> f16 [dir][n][H], n = 2p (re) / 2p+1 (im) ============ */
__global__ void s5_bb(const float* __restrict__ Bre, const float* __restrict__ Bim,
                      const float* __restrict__ wbuf, _Float16* __restrict__ bb16) {
  int t = blockIdx.x * blockDim.x + threadIdx.x;          /* 2*P*H = 262144 */
  int h = t % HDIM;
  int p = (t / HDIM) % PDIM;
  int dir = t / (HDIM * PDIM);
  float br = Bre[t], bi = Bim[t];
  float wr = wbuf[(dir * 2 + 0) * PDIM + p];
  float wi = wbuf[(dir * 2 + 1) * PDIM + p];
  size_t base = (size_t)dir * K2P * HDIM;
  bb16[base + (size_t)(2 * p + 0) * HDIM + h] = (_Float16)(wr * br - wi * bi);
  bb16[base + (size_t)(2 * p + 1) * HDIM + h] = (_Float16)(wr * bi + wi * br);
}

/* ============ C -> f16 [dir][H][k], k = 2p (re) / 2p+1 (-im); D sum ============ */
__global__ void s5_ct(const float* __restrict__ Cre, const float* __restrict__ Cim,
                      const float* __restrict__ Dd,
                      _Float16* __restrict__ ct16, float* __restrict__ dsum) {
  int t = blockIdx.x * blockDim.x + threadIdx.x;          /* 2*H*P = 262144 */
  int p = t % PDIM;
  int h = (t / PDIM) % HDIM;
  int dir = t / (PDIM * HDIM);
  size_t base = (size_t)dir * HDIM * K2P + (size_t)h * K2P;
  ct16[base + 2 * p + 0] = (_Float16)Cre[t];
  ct16[base + 2 * p + 1] = (_Float16)(-Cim[t]);           /* Re(s*c) = sr*cr - si*ci */
  if (t < HDIM) dsum[t] = Dd[t] + Dd[HDIM + t];
}

/* ============ x fp32 -> f16 ============ */
__global__ void s5_x16(const float* __restrict__ x, _Float16* __restrict__ x16) {
  size_t i8 = ((size_t)blockIdx.x * blockDim.x + threadIdx.x) * 8;
  v4f a = *(const v4f*)(x + i8);
  v4f b = *(const v4f*)(x + i8 + 4);
  v8h o;
  o[0] = (_Float16)a[0]; o[1] = (_Float16)a[1]; o[2] = (_Float16)a[2]; o[3] = (_Float16)a[3];
  o[4] = (_Float16)b[0]; o[5] = (_Float16)b[1]; o[6] = (_Float16)b[2]; o[7] = (_Float16)b[3];
  *(v8h*)(x16 + i8) = o;
}

/* two 16-byte chunks -> one v16h WMMA operand (works for global or LDS after inlining) */
__device__ __forceinline__ v16h ld16(const _Float16* lo, const _Float16* hi) {
  v8h a = *(const v8h*)lo;
  v8h b = *(const v8h*)hi;
  return __builtin_shufflevector(a, b, 0, 1, 2, 3, 4, 5, 6, 7,
                                 8, 9, 10, 11, 12, 13, 14, 15);
}

/* ===== GEMM1: bu[dir][m][2P] = x16[m][H] * bb16[dir][2P][H]^T ===============
   Block = one (dir, m-tile); A tile staged in LDS (shared by 8 waves);
   each wave computes 4 N-tiles (64 cols). ======================================= */
__global__ void s5_gemm1(const _Float16* __restrict__ x16,
                         const _Float16* __restrict__ bb16,
                         _Float16* __restrict__ bu16) {
  __shared__ _Float16 Atile[16 * ROWP];                    /* 16.6 KB, padded rows */
  int tid = threadIdx.x;
  int mt  = blockIdx.x & 2047;
  int dir = blockIdx.x >> 11;                              /* 4096 blocks total */
  /* cooperative stage: row = tid/16, 32-half slice = (tid%16)*32 */
  {
    int r = tid >> 4;
    int s = (tid & 15) * 32;
    const _Float16* src = x16 + (size_t)(mt * 16 + r) * HDIM + s;
    _Float16* dst = Atile + (size_t)r * ROWP + s;
    *(v8h*)(dst + 0)  = *(const v8h*)(src + 0);
    *(v8h*)(dst + 8)  = *(const v8h*)(src + 8);
    *(v8h*)(dst + 16) = *(const v8h*)(src + 16);
    *(v8h*)(dst + 24) = *(const v8h*)(src + 24);
  }
  __syncthreads();
  int lane = tid & 31;
  int ntg  = tid >> 5;                                     /* wave id = N group (8 x 64 cols) */
  int lm = lane & 15, sel = lane >> 4;
  const _Float16* arow  = Atile + (size_t)lm * ROWP;
  const _Float16* bbase = bb16 + (size_t)dir * K2P * HDIM + (size_t)(ntg * 64 + lm) * HDIM;
  v8f acc[4] = {{}, {}, {}, {}};
#pragma unroll
  for (int kt = 0; kt < HDIM / 32; ++kt) {                 /* 16 * 4 = 64 WMMAs */
    int k0 = kt * 32;
    v16h A = ld16(arow + k0 + sel * 8, arow + k0 + sel * 8 + 16);
#pragma unroll
    for (int u = 0; u < 4; ++u) {
      const _Float16* brow = bbase + (size_t)(u * 16) * HDIM;
      v16h B = ld16(brow + k0 + sel * 8, brow + k0 + sel * 8 + 16);
      acc[u] = __builtin_amdgcn_wmma_f32_16x16x32_f16(false, A, false, B,
                                                      (short)0, acc[u], false, false);
    }
  }
  _Float16* out = bu16 + (size_t)dir * BL * K2P;
  int ncol = ntg * 64 + lm;
#pragma unroll
  for (int r = 0; r < 8; ++r) {
    int m = mt * 16 + r + sel * 8;                         /* C/D layout: M = r + 8*sel */
    _Float16* orow = out + (size_t)m * K2P + ncol;
#pragma unroll
    for (int u = 0; u < 4; ++u) orow[u * 16] = (_Float16)acc[u][r];
  }
}

/* ============ blocked scan, pass 1: per-chunk local sum S ============ */
__global__ void s5_scan1(const _Float16* __restrict__ bu16,
                         const float* __restrict__ abar,
                         float* __restrict__ Sre, float* __restrict__ Sim) {
  int t = blockIdx.x * blockDim.x + threadIdx.x;            /* 2*4*16*256 = 32768 */
  int p = t & (PDIM - 1);
  int c = (t / PDIM) & (NCHUNK - 1);
  int b = (t / (PDIM * NCHUNK)) & 3;
  int dir = t / (PDIM * NCHUNK * 4);
  float ar = abar[(dir * 2 + 0) * PDIM + p];
  float ai = abar[(dir * 2 + 1) * PDIM + p];
  const _Float16* base = bu16 + ((size_t)dir * BL + (size_t)b * SEQL + (size_t)c * CHUNK) * K2P
                         + 2 * p;
  float sr = 0.f, si = 0.f;
  if (dir == 0) {
    for (int j = 0; j < CHUNK; ++j) {
      __builtin_prefetch((const void*)(base + (size_t)(j + PFD) * K2P), 0, 1);
      h2 v = *(const h2*)(base + (size_t)j * K2P);
      float nr = ar * sr - ai * si + (float)v[0];
      float ni = ar * si + ai * sr + (float)v[1];
      sr = nr; si = ni;
    }
  } else {
    for (int j = CHUNK - 1; j >= 0; --j) {
      __builtin_prefetch((const void*)(base + (size_t)(j - PFD) * K2P), 0, 1);
      h2 v = *(const h2*)(base + (size_t)j * K2P);
      float nr = ar * sr - ai * si + (float)v[0];
      float ni = ar * si + ai * sr + (float)v[1];
      sr = nr; si = ni;
    }
  }
  Sre[t] = sr; Sim[t] = si;
}

/* ============ blocked scan, pass 2: carry across chunks ============ */
__global__ void s5_scan2(const float* __restrict__ abar,
                         const float* __restrict__ Sre, const float* __restrict__ Sim,
                         float* __restrict__ Cr, float* __restrict__ Ci) {
  int t = blockIdx.x * blockDim.x + threadIdx.x;            /* 2*4*256 = 2048 */
  int p = t & (PDIM - 1);
  int b = (t / PDIM) & 3;
  int dir = t / (PDIM * 4);
  /* aCH = a^CHUNK via 9 squarings (CHUNK = 2^9) */
  float pr = abar[(dir * 2 + 0) * PDIM + p];
  float pi = abar[(dir * 2 + 1) * PDIM + p];
#pragma unroll
  for (int i = 0; i < 9; ++i) {
    float nr = pr * pr - pi * pi;
    float ni = 2.0f * pr * pi;
    pr = nr; pi = ni;
  }
  size_t row = ((size_t)dir * 4 + b) * NCHUNK;
  float cr = 0.f, ci = 0.f;
  if (dir == 0) {
    for (int c = 0; c < NCHUNK; ++c) {
      size_t idx = (row + c) * PDIM + p;
      Cr[idx] = cr; Ci[idx] = ci;
      float nr = pr * cr - pi * ci + Sre[idx];
      float ni = pr * ci + pi * cr + Sim[idx];
      cr = nr; ci = ni;
    }
  } else {
    for (int c = NCHUNK - 1; c >= 0; --c) {
      size_t idx = (row + c) * PDIM + p;
      Cr[idx] = cr; Ci[idx] = ci;
      float nr = pr * cr - pi * ci + Sre[idx];
      float ni = pr * ci + pi * cr + Sim[idx];
      cr = nr; ci = ni;
    }
  }
}

/* ============ blocked scan, pass 3: replay with carry, write states in-place ============ */
__global__ void s5_scan3(_Float16* __restrict__ bu16,
                         const float* __restrict__ abar,
                         const float* __restrict__ Cr, const float* __restrict__ Ci) {
  int t = blockIdx.x * blockDim.x + threadIdx.x;            /* 32768 */
  int p = t & (PDIM - 1);
  int c = (t / PDIM) & (NCHUNK - 1);
  int b = (t / (PDIM * NCHUNK)) & 3;
  int dir = t / (PDIM * NCHUNK * 4);
  float ar = abar[(dir * 2 + 0) * PDIM + p];
  float ai = abar[(dir * 2 + 1) * PDIM + p];
  _Float16* base = bu16 + ((size_t)dir * BL + (size_t)b * SEQL + (size_t)c * CHUNK) * K2P
                   + 2 * p;
  float sr = Cr[t], si = Ci[t];
  if (dir == 0) {
    for (int j = 0; j < CHUNK; ++j) {
      __builtin_prefetch((const void*)(base + (size_t)(j + PFD) * K2P), 0, 1);
      h2* q = (h2*)(base + (size_t)j * K2P);
      h2 v = *q;
      float nr = ar * sr - ai * si + (float)v[0];
      float ni = ar * si + ai * sr + (float)v[1];
      sr = nr; si = ni;
      h2 o; o[0] = (_Float16)sr; o[1] = (_Float16)si;
      *q = o;
    }
  } else {
    for (int j = CHUNK - 1; j >= 0; --j) {
      __builtin_prefetch((const void*)(base + (size_t)(j - PFD) * K2P), 0, 1);
      h2* q = (h2*)(base + (size_t)j * K2P);
      h2 v = *q;
      float nr = ar * sr - ai * si + (float)v[0];
      float ni = ar * si + ai * sr + (float)v[1];
      sr = nr; si = ni;
      h2 o; o[0] = (_Float16)sr; o[1] = (_Float16)si;
      *q = o;
    }
  }
}

/* ===== GEMM2: y[m][H] = sum_dir states[dir][m][2P]*ct[dir][H][2P]^T + x*Dsum ====
   Block = one m-tile; both directions' A rows staged in LDS; 4 N-tiles/wave. ==== */
__global__ void s5_gemm2(const _Float16* __restrict__ st16,
                         const _Float16* __restrict__ ct16,
                         const float* __restrict__ x,
                         const float* __restrict__ dsum,
                         float* __restrict__ y) {
  __shared__ _Float16 Atile[32 * ROWP];                    /* 2 dirs x 16 rows, 33 KB */
  int tid = threadIdx.x;
  int mt  = blockIdx.x;                                    /* 2048 blocks */
  /* cooperative stage: 32 rows; row = tid/8, 64-half slice = (tid%8)*64 */
  {
    int row = tid >> 3;
    int d   = row >> 4;
    int r   = row & 15;
    int s   = (tid & 7) * 64;
    const _Float16* src = st16 + ((size_t)d * BL + (size_t)(mt * 16 + r)) * K2P + s;
    _Float16* dst = Atile + (size_t)row * ROWP + s;
#pragma unroll
    for (int q = 0; q < 8; ++q)
      *(v8h*)(dst + q * 8) = *(const v8h*)(src + q * 8);
  }
  __syncthreads();
  int lane = tid & 31;
  int ntg  = tid >> 5;                                     /* wave id = H group (8 x 64 cols) */
  int lm = lane & 15, sel = lane >> 4;
  v8f acc[4] = {{}, {}, {}, {}};
#pragma unroll
  for (int dir = 0; dir < 2; ++dir) {
    const _Float16* arow  = Atile + (size_t)(dir * 16 + lm) * ROWP;
    const _Float16* bbase = ct16 + (size_t)dir * HDIM * K2P + (size_t)(ntg * 64 + lm) * K2P;
#pragma unroll
    for (int kt = 0; kt < K2P / 32; ++kt) {                /* 2*16*4 = 128 WMMAs */
      int k0 = kt * 32;
      v16h A = ld16(arow + k0 + sel * 8, arow + k0 + sel * 8 + 16);
#pragma unroll
      for (int u = 0; u < 4; ++u) {
        const _Float16* brow = bbase + (size_t)(u * 16) * K2P;
        v16h B = ld16(brow + k0 + sel * 8, brow + k0 + sel * 8 + 16);
        acc[u] = __builtin_amdgcn_wmma_f32_16x16x32_f16(false, A, false, B,
                                                        (short)0, acc[u], false, false);
      }
    }
  }
  int hb = ntg * 64 + lm;
#pragma unroll
  for (int u = 0; u < 4; ++u) {
    int h = hb + u * 16;
    float d = dsum[h];
#pragma unroll
    for (int r = 0; r < 8; ++r) {
      size_t m = (size_t)(mt * 16 + r + sel * 8);
      size_t idx = m * HDIM + h;
      y[idx] = acc[u][r] + x[idx] * d;
    }
  }
}

extern "C" void kernel_launch(void* const* d_in, const int* in_sizes, int n_in,
                              void* d_out, int out_size, void* d_ws, size_t ws_size,
                              hipStream_t stream) {
  const float* x   = (const float*)d_in[0];
  const float* lre = (const float*)d_in[1];
  const float* lim = (const float*)d_in[2];
  const float* ldt = (const float*)d_in[3];
  const float* Bre = (const float*)d_in[4];
  const float* Bim = (const float*)d_in[5];
  const float* Cre = (const float*)d_in[6];
  const float* Cim = (const float*)d_in[7];
  const float* Dd  = (const float*)d_in[8];

  char* ws = (char*)d_ws;
  _Float16* x16  = (_Float16*)(ws + OFF_X16);
  _Float16* bu16 = (_Float16*)(ws + OFF_BU);
  _Float16* bb16 = (_Float16*)(ws + OFF_BB);
  _Float16* ct16 = (_Float16*)(ws + OFF_CT);
  float* abar = (float*)(ws + OFF_ABAR);
  float* wbuf = (float*)(ws + OFF_W);
  float* dsum = (float*)(ws + OFF_DSUM);
  float* Sre  = (float*)(ws + OFF_SCAN);
  float* Sim  = Sre + 32768;
  float* Cr   = Sim + 32768;
  float* Ci   = Cr + 32768;
  float* y = (float*)d_out;

  s5_params<<<2,    256, 0, stream>>>(lre, lim, ldt, abar, wbuf);
  s5_bb   <<<1024,  256, 0, stream>>>(Bre, Bim, wbuf, bb16);
  s5_ct   <<<1024,  256, 0, stream>>>(Cre, Cim, Dd, ct16, dsum);
  s5_x16  <<<8192,  256, 0, stream>>>(x, x16);
  s5_gemm1<<<4096,  256, 0, stream>>>(x16, bb16, bu16);
  s5_scan1<<<128,   256, 0, stream>>>(bu16, abar, Sre, Sim);
  s5_scan2<<<8,     256, 0, stream>>>(abar, Sre, Sim, Cr, Ci);
  s5_scan3<<<128,   256, 0, stream>>>(bu16, abar, Cr, Ci);
  s5_gemm2<<<2048,  256, 0, stream>>>(bu16, ct16, x, dsum, y);
}
